// HETXLHead_63556926046817
// MI455X (gfx1250) — compile-verified, matrix operand order checked by reading, not run on hardware
//
#include <hip/hip_runtime.h>
#include <hip/hip_bf16.h>
#include <stdint.h>

typedef __attribute__((ext_vector_type(16))) __bf16 v16bf;
typedef __attribute__((ext_vector_type(8)))  __bf16 bf16x8;
typedef __attribute__((ext_vector_type(4)))  __bf16 bf16x4;
typedef __attribute__((ext_vector_type(8)))  float  v8f;
typedef __attribute__((ext_vector_type(4)))  int    v4i;

#define TEMPERATURE_INV (1.0f / 1.5f)
#define MIN_SCALE 1e-3f

enum { EPI_BF16 = 0, EPI_SOFTPLUS = 1, EPI_SCALE = 2 };

// Block tile: 128x128x32, 256 threads = 8 wave32 (4 in M x 2 in N),
// each wave: 2x4 grid of 16x16 WMMA tiles, double-buffered LDS.
constexpr int BM = 128, BN = 128, BK = 32;
constexpr int LDST = 40;   // bf16 elems per LDS row (80 B, 16B-aligned, conflict-free)

#if defined(__gfx1250__) && __has_builtin(__builtin_amdgcn_global_load_async_to_lds_b128) && __has_builtin(__builtin_amdgcn_s_wait_asynccnt)
#define HAVE_ASYNC_LDS 1
#else
#define HAVE_ASYNC_LDS 0
#endif

static __device__ inline v16bf combine8(bf16x8 lo, bf16x8 hi) {
  return __builtin_shufflevector(lo, hi, 0,1,2,3,4,5,6,7,8,9,10,11,12,13,14,15);
}

#if HAVE_ASYNC_LDS
static __device__ inline void async_copy_b128(const void* g, void* l) {
  auto gp = (__attribute__((address_space(1))) v4i*)(uintptr_t)g;
  auto lp = (__attribute__((address_space(3))) v4i*)(uint32_t)(uintptr_t)l;
  __builtin_amdgcn_global_load_async_to_lds_b128(gp, lp, 0, 0);
}
#endif

// C[M,N] = A[M,K] @ W[N,K]^T + bias  (A,W f32 or bf16; bf16 WMMA, f32 accum)
// All tile dims are exact multiples (W_cls zero-padded to 1024 rows), so the
// staging loops are completely unguarded straight-line vector loads.
template <bool A_BF16, bool B_BF16, int EPI>
__global__ void __launch_bounds__(256)
gemm_wmma(const void* __restrict__ Av, const void* __restrict__ Wv,
          const float* __restrict__ bias, void* __restrict__ outv,
          int K, int Nact, int ldOut)
{
  __shared__ __attribute__((aligned(16))) __bf16 lA[2][BM * LDST];
  __shared__ __attribute__((aligned(16))) __bf16 lB[2][BN * LDST];

  const int tid     = threadIdx.x;
  const int lane    = tid & 31;
  const int wave    = tid >> 5;
  const int wm      = wave & 3;            // 32-row slab within block tile
  const int wn      = wave >> 2;           // 64-col slab within block tile
  const int laneRow = lane & 15;
  const int off     = (lane < 16) ? 0 : 8; // ISA 16-bit A/B lane K-offset & C/D row offset

  const int64_t rowBase = (int64_t)blockIdx.y * BM;
  const int64_t colBase = (int64_t)blockIdx.x * BN;
  const int KT = K / BK;

  v8f acc[2][4] = {};

  // ---- per-thread staging coordinates ----
  const int rH = tid >> 2, cH = (tid & 3) << 3;        // bf16 source: 2 chunks/thread
  const int rF = tid >> 3, cF = (tid & 7) << 2;        // f32  source: 4 chunks/thread

  auto computeTile = [&](const __bf16* bA, const __bf16* bB) {
    v16bf fa[2], fb[4];
    for (int tm = 0; tm < 2; ++tm) {
      const __bf16* p = &bA[(wm * 32 + tm * 16 + laneRow) * LDST + off];
      fa[tm] = combine8(*(const bf16x8*)p, *(const bf16x8*)(p + 16));
    }
    for (int tn = 0; tn < 4; ++tn) {
      const __bf16* p = &bB[(wn * 64 + tn * 16 + laneRow) * LDST + off];
      fb[tn] = combine8(*(const bf16x8*)p, *(const bf16x8*)(p + 16));
    }
    for (int tm = 0; tm < 2; ++tm)
      for (int tn = 0; tn < 4; ++tn)
        acc[tm][tn] = __builtin_amdgcn_wmma_f32_16x16x32_bf16(
            false, fa[tm], false, fb[tn], (short)0, acc[tm][tn], false, false);
  };

#if HAVE_ASYNC_LDS
  if (A_BF16 && B_BF16) {
    // --------- async-to-LDS pipeline (no conversion needed) ---------
    const __bf16* A = (const __bf16*)Av;
    const __bf16* W = (const __bf16*)Wv;
    auto stage = [&](int buf, int k0) {   // 4 async ops per wave per tile
      for (int it = 0; it < 2; ++it) {
        int r = rH + it * 64, c = cH;
        async_copy_b128(A + (rowBase + r) * (int64_t)K + k0 + c, &lA[buf][r * LDST + c]);
        async_copy_b128(W + (colBase + r) * (int64_t)K + k0 + c, &lB[buf][r * LDST + c]);
      }
    };
    stage(0, 0);
    for (int kt = 0; kt < KT; ++kt) {
      const int cur = kt & 1;
      if (kt + 1 < KT) {
        stage(cur ^ 1, (kt + 1) * BK);               // overlap with compute of kt
        __builtin_amdgcn_s_wait_asynccnt(4);         // tile kt's 4 copies done
      } else {
        __builtin_amdgcn_s_wait_asynccnt(0);
      }
      __syncthreads();                               // all waves' copies visible
      computeTile(lA[cur], lB[cur]);
      __syncthreads();                               // protect cur from next async writes
    }
  } else
#endif
  {
    // --------- register-prefetch double-buffered pipeline ---------
    bf16x8 ha[2], hb[2];
    float4 fa4[4], fb4[4];
    auto loadRegs = [&](int k0) {
      if (A_BF16) {
        const __bf16* A = (const __bf16*)Av;
        for (int it = 0; it < 2; ++it)
          ha[it] = *(const bf16x8*)(A + (rowBase + rH + it * 64) * (int64_t)K + k0 + cH);
      } else {
        const float* A = (const float*)Av;
        for (int it = 0; it < 4; ++it)
          fa4[it] = *(const float4*)(A + (rowBase + rF + it * 32) * (int64_t)K + k0 + cF);
      }
      if (B_BF16) {
        const __bf16* W = (const __bf16*)Wv;
        for (int it = 0; it < 2; ++it)
          hb[it] = *(const bf16x8*)(W + (colBase + rH + it * 64) * (int64_t)K + k0 + cH);
      } else {
        const float* W = (const float*)Wv;
        for (int it = 0; it < 4; ++it)
          fb4[it] = *(const float4*)(W + (colBase + rF + it * 32) * (int64_t)K + k0 + cF);
      }
    };
    auto storeLDS = [&](int buf) {
      if (A_BF16) {
        for (int it = 0; it < 2; ++it)
          *(bf16x8*)&lA[buf][(rH + it * 64) * LDST + cH] = ha[it];
      } else {
        for (int it = 0; it < 4; ++it) {
          bf16x4 h;
          h[0] = (__bf16)fa4[it].x; h[1] = (__bf16)fa4[it].y;
          h[2] = (__bf16)fa4[it].z; h[3] = (__bf16)fa4[it].w;
          *(bf16x4*)&lA[buf][(rF + it * 32) * LDST + cF] = h;
        }
      }
      if (B_BF16) {
        for (int it = 0; it < 2; ++it)
          *(bf16x8*)&lB[buf][(rH + it * 64) * LDST + cH] = hb[it];
      } else {
        for (int it = 0; it < 4; ++it) {
          bf16x4 h;
          h[0] = (__bf16)fb4[it].x; h[1] = (__bf16)fb4[it].y;
          h[2] = (__bf16)fb4[it].z; h[3] = (__bf16)fb4[it].w;
          *(bf16x4*)&lB[buf][(rF + it * 32) * LDST + cF] = h;
        }
      }
    };

    loadRegs(0);
    storeLDS(0);
    __syncthreads();
    for (int kt = 0; kt < KT; ++kt) {
      const int cur = kt & 1;
      if (kt + 1 < KT) loadRegs((kt + 1) * BK);      // issue next-tile loads early
      if (kt + 2 < KT) {                             // L2 prefetch two tiles ahead
        const char* pa = A_BF16
            ? (const char*)((const __bf16*)Av + (rowBase + rH) * (int64_t)K + (kt + 2) * BK)
            : (const char*)((const float*)Av + (rowBase + rF) * (int64_t)K + (kt + 2) * BK);
        __builtin_prefetch(pa, 0, 0);
        const char* pb = B_BF16
            ? (const char*)((const __bf16*)Wv + (colBase + rH) * (int64_t)K + (kt + 2) * BK)
            : (const char*)((const float*)Wv + (colBase + rF) * (int64_t)K + (kt + 2) * BK);
        __builtin_prefetch(pb, 0, 0);
      }
      computeTile(lA[cur], lB[cur]);
      if (kt + 1 < KT) {
        storeLDS(cur ^ 1);
        __syncthreads();
      }
    }
  }

  // ---- epilogue (C/D layout: VGPR j -> row j + off, col = laneRow) ----
  for (int tm = 0; tm < 2; ++tm) {
    for (int tn = 0; tn < 4; ++tn) {
      int64_t gc = colBase + wn * 64 + tn * 16 + laneRow;
      if (gc >= Nact) continue;
      float bsum = bias[gc];
      for (int j = 0; j < 8; ++j) {
        int64_t gr = rowBase + wm * 32 + tm * 16 + off + j;
        float x = acc[tm][tn][j] + bsum;
        if (EPI == EPI_BF16) {
          ((__bf16*)outv)[gr * (int64_t)ldOut + gc] = (__bf16)x;
        } else if (EPI == EPI_SOFTPLUS) {
          float sp = (x > 20.f) ? x : log1pf(__expf(x));
          ((float*)outv)[gr * (int64_t)ldOut + gc] = sp + MIN_SCALE;
        } else { // EPI_SCALE: streamed output, bypass caches
          __builtin_nontemporal_store(x * TEMPERATURE_INV,
                                      &((float*)outv)[gr * (int64_t)ldOut + gc]);
        }
      }
    }
  }
}

// pre_logits(b,s,d) = feat(b,d) + std(b,d)*nd(b,s,d) + sum_r lrc(b,d,r)*nlr(b,s,r)
__global__ void __launch_bounds__(256)
build_prelogits(const float* __restrict__ feat, const float* __restrict__ stdv,
                const __bf16* __restrict__ lrc, const float* __restrict__ nlr,
                const float* __restrict__ nd, __bf16* __restrict__ out)
{
  const int bs = blockIdx.x;   // b*64 + s
  const int b  = bs >> 6;
  const int tid = threadIdx.x;
  __shared__ float snlr[16];
  if (tid < 16) snlr[tid] = nlr[(int64_t)bs * 16 + tid];
  __syncthreads();
  for (int i = 0; i < 2; ++i) {
    int d = tid + i * 256;
    const __bf16* lp = lrc + ((int64_t)b * 512 + d) * 16;
    bf16x8 l0 = *(const bf16x8*)(lp);
    bf16x8 l1 = *(const bf16x8*)(lp + 8);
    float ndv = __builtin_nontemporal_load(&nd[(int64_t)bs * 512 + d]);
    float s = feat[(int64_t)b * 512 + d] + stdv[(int64_t)b * 512 + d] * ndv;
    for (int r = 0; r < 8; ++r)
      s += (float)l0[r] * snlr[r] + (float)l1[r] * snlr[r + 8];
    __builtin_nontemporal_store((__bf16)s, &out[(int64_t)bs * 512 + d]);
  }
}

// W_cls (1000x512 f32) -> bf16, zero-padded to 1024 rows so the big GEMM
// needs no staging guards at all.
__global__ void __launch_bounds__(256)
wcls_to_bf16_pad(const float* __restrict__ in, __bf16* __restrict__ out, int nIn, int nPad)
{
  int i = (blockIdx.x * 256 + threadIdx.x) * 4;
  if (i >= nPad) return;
  bf16x4 h = {};
  if (i < nIn) {
    float4 v = *(const float4*)(in + i);
    h[0] = (__bf16)v.x; h[1] = (__bf16)v.y; h[2] = (__bf16)v.z; h[3] = (__bf16)v.w;
  }
  *(bf16x4*)(out + i) = h;
}

extern "C" void kernel_launch(void* const* d_in, const int* in_sizes, int n_in,
                              void* d_out, int out_size, void* d_ws, size_t ws_size,
                              hipStream_t stream) {
  const float* features   = (const float*)d_in[0];
  const float* W_cov      = (const float*)d_in[1];
  const float* b_cov      = (const float*)d_in[2];
  const float* W_diag     = (const float*)d_in[3];
  const float* b_diag     = (const float*)d_in[4];
  const float* W_cls      = (const float*)d_in[5];
  const float* b_cls      = (const float*)d_in[6];
  const float* noise_diag = (const float*)d_in[7];
  const float* noise_lr   = (const float*)d_in[8];

  const int B = 1024, S = 64, D = 512, R = 16, C = 1000;
  const int DR = D * R;            // 8192
  const int Cpad = 1024;           // C padded to 128-multiple for GEMM tiling

  // workspace layout
  char* ws = (char*)d_ws;
  __bf16* lrc_bf16  = (__bf16*)(ws);                                        // 16.78 MB
  float*  std_f32   = (float*)(ws + (size_t)16777216);                      //  2.10 MB
  __bf16* wcls_bf16 = (__bf16*)(ws + (size_t)16777216 + 2097152);           //  1.05 MB (padded)
  __bf16* pre_bf16  = (__bf16*)(ws + (size_t)16777216 + 2097152 + 1048576); // 67.1 MB

  // 1) low_rank_cov (B x D*R), stored bf16 row-major == reshape(B,D,R)
  gemm_wmma<false, false, EPI_BF16><<<dim3(DR / BN, B / BM), 256, 0, stream>>>(
      features, W_cov, b_cov, lrc_bf16, D, DR, DR);

  // 2) diagonal_std (B x D), fused softplus + MIN_SCALE
  gemm_wmma<false, false, EPI_SOFTPLUS><<<dim3(D / BN, B / BM), 256, 0, stream>>>(
      features, W_diag, b_diag, std_f32, D, D, D);

  // 3) W_cls -> bf16, zero-padded to 1024 rows
  wcls_to_bf16_pad<<<(Cpad * D / 4 + 255) / 256, 256, 0, stream>>>(
      W_cls, wcls_bf16, C * D, Cpad * D);

  // 4) fused samples + feature add -> pre_logits (B*S x D) bf16
  build_prelogits<<<B * S, 256, 0, stream>>>(features, std_f32, lrc_bf16,
                                             noise_lr, noise_diag, pre_bf16);

  // 5) logits = pre_logits @ W_cls^T, fused bias + 1/TEMPERATURE, f32 out
  gemm_wmma<true, true, EPI_SCALE><<<dim3(Cpad / BN, (B * S) / BM), 256, 0, stream>>>(
      pre_bf16, wcls_bf16, b_cls, d_out, D, C, C);
}
